// S_decoder_5583457485492
// MI455X (gfx1250) — matrix operand order repaired and checked by reading, not compiled, not run on hardware
//
#include <hip/hip_runtime.h>
#include <math.h>

typedef _Float16 v16h __attribute__((ext_vector_type(16)));
typedef float    v8f  __attribute__((ext_vector_type(8)));

#define DIN  64
#define DHID 128

// ---------------------------------------------------------------------------
// Degree / normalization
// ---------------------------------------------------------------------------
__global__ void fill_deg(float* __restrict__ deg, int n) {
    int i = blockIdx.x * blockDim.x + threadIdx.x;
    if (i < n) deg[i] = 1.0f;            // self-loop contributes 1 to every node
}

__global__ void deg_scatter(const int* __restrict__ edges, float* __restrict__ deg, int E) {
    int e = blockIdx.x * blockDim.x + threadIdx.x;
    if (e < E) atomicAdd(deg + edges[E + e], 1.0f);   // col = edge_index[1]
}

__global__ void calc_dinv(const float* __restrict__ deg, float* __restrict__ dinv, int n) {
    int i = blockIdx.x * blockDim.x + threadIdx.x;
    if (i < n) dinv[i] = rsqrtf(deg[i]);              // deg >= 1 always
}

// ---------------------------------------------------------------------------
// Layer-1 dense GEMM: xw = |x| @ W1   (M=n, K=64, N=128) via WMMA f16->f32
// One wave per 16-row M-tile; 8 accumulator tiles cover all 128 output cols.
// ---------------------------------------------------------------------------
__global__ __launch_bounds__(256)
void gemm1_wmma(const float* __restrict__ x, const float* __restrict__ W1,
                float* __restrict__ xw, int n) {
    // Stage W1 transposed as f16:  sW[col][k], row length 64 halves (128B)
    __shared__ __attribute__((aligned(32))) _Float16 sW[DHID * DIN];
    const int tid = threadIdx.x;
    for (int i = tid; i < DHID * DIN; i += 256) {
        int nn = i >> 6;          // output col 0..127
        int kk = i & 63;          // k 0..63
        sW[i] = (_Float16)W1[kk * DHID + nn];
    }
    __syncthreads();

    const int lane  = tid & 31;
    const int wave  = tid >> 5;
    const int mbase = (blockIdx.x * 8 + wave) * 16;
    if (mbase >= n) return;                        // wave-uniform (n % 16 == 0)

    const int m  = mbase + (lane & 15);
    const int hi = (lane & 16) ? 1 : 0;

    v8f zero = {};
    v8f acc[8];
#pragma unroll
    for (int t = 0; t < 8; ++t) acc[t] = zero;

#pragma unroll
    for (int kc = 0; kc < 2; ++kc) {
        // A fragment (16x32 f16): lanes 0-15 hold K kc*32+{0..7,16..23},
        //                         lanes 16-31 hold K kc*32+{8..15,24..31}
        const int k0 = kc * 32 + hi * 8;
        const float* xr = x + (size_t)m * DIN + k0;
        float4 p0 = *(const float4*)(xr +  0);
        float4 p1 = *(const float4*)(xr +  4);
        float4 p2 = *(const float4*)(xr + 16);
        float4 p3 = *(const float4*)(xr + 20);
        v16h a;
        a[0]  = (_Float16)fabsf(p0.x); a[1]  = (_Float16)fabsf(p0.y);
        a[2]  = (_Float16)fabsf(p0.z); a[3]  = (_Float16)fabsf(p0.w);
        a[4]  = (_Float16)fabsf(p1.x); a[5]  = (_Float16)fabsf(p1.y);
        a[6]  = (_Float16)fabsf(p1.z); a[7]  = (_Float16)fabsf(p1.w);
        a[8]  = (_Float16)fabsf(p2.x); a[9]  = (_Float16)fabsf(p2.y);
        a[10] = (_Float16)fabsf(p2.z); a[11] = (_Float16)fabsf(p2.w);
        a[12] = (_Float16)fabsf(p3.x); a[13] = (_Float16)fabsf(p3.y);
        a[14] = (_Float16)fabsf(p3.z); a[15] = (_Float16)fabsf(p3.w);

        // B fragment (32x16 f16): lane = col, lanes 0-15 K=kb+{0..7,16..23},
        //                         lanes 16-31 K=kc*32+{8..15,24..31} (kb includes hi*8)
        const int kb = kc * 32 + hi * 8;
#pragma unroll
        for (int t = 0; t < 8; ++t) {
            typedef _Float16 v8h_t __attribute__((ext_vector_type(8)));
            const _Float16* bp = &sW[(t * 16 + (lane & 15)) * DIN + kb];
            v8h_t b0 = *(const v8h_t*)(bp);
            v8h_t b1v = *(const v8h_t*)(bp + 16);
            v16h b = __builtin_shufflevector(b0, b1v, 0,1,2,3,4,5,6,7,8,9,10,11,12,13,14,15);
            acc[t] = __builtin_amdgcn_wmma_f32_16x16x32_f16(
                false, a, false, b, (short)0, acc[t], false, false);
        }
    }

    // D layout: lanes 0-15 -> N=lane, M=r; lanes 16-31 -> N=lane-16, M=r+8
    const int mrow = mbase + (hi ? 8 : 0);
    const int ncol = lane & 15;
#pragma unroll
    for (int t = 0; t < 8; ++t)
#pragma unroll
        for (int r = 0; r < 8; ++r)
            xw[(size_t)(mrow + r) * DHID + t * 16 + ncol] = acc[t][r];
}

// ---------------------------------------------------------------------------
// agg init = self-loop message:  agg[i][:] = dinv[i]^2 * xw[i][:]
// ---------------------------------------------------------------------------
__global__ void init_agg(const float* __restrict__ xw, const float* __restrict__ dinv,
                         float* __restrict__ agg, int n) {
    int i = blockIdx.x * blockDim.x + threadIdx.x;   // float4 index
    if (i >= n * 32) return;
    int node = i >> 5;
    float w = dinv[node]; w *= w;
    float4 v = ((const float4*)xw)[i];
    float4 o = {w * v.x, w * v.y, w * v.z, w * v.w};
    ((float4*)agg)[i] = o;
}

// ---------------------------------------------------------------------------
// Edge scatter, layer 1: one wave per edge, lane l moves a float4 of the msg
// ---------------------------------------------------------------------------
__global__ __launch_bounds__(256)
void scatter1(const int* __restrict__ edges, const float* __restrict__ dinv,
              const float* __restrict__ xw, float* __restrict__ agg, int E) {
    const int lane = threadIdx.x & 31;
    const int nw   = (gridDim.x * blockDim.x) >> 5;
    for (int e = (blockIdx.x * blockDim.x + threadIdx.x) >> 5; e < E; e += nw) {
        const int r = edges[e];
        const int c = edges[E + e];
        const float w = dinv[r] * dinv[c];
        const float4 v = *(const float4*)(xw + (size_t)r * DHID + 4 * lane);
        float* dst = agg + (size_t)c * DHID + 4 * lane;
        atomicAdd(dst + 0, w * v.x);
        atomicAdd(dst + 1, w * v.y);
        atomicAdd(dst + 2, w * v.z);
        atomicAdd(dst + 3, w * v.w);
    }
}

// ---------------------------------------------------------------------------
// Layer 2 start: z[i] = sum_d relu(agg[i][d] + b1[d]) * W2[d]   (wave GEMV)
// ---------------------------------------------------------------------------
__global__ __launch_bounds__(256)
void gemv2(const float* __restrict__ agg, const float* __restrict__ b1,
           const float* __restrict__ W2, float* __restrict__ z, int n) {
    const int node = blockIdx.x * 8 + (threadIdx.x >> 5);
    const int lane = threadIdx.x & 31;
    if (node >= n) return;
    const float4 a = *(const float4*)(agg + (size_t)node * DHID + 4 * lane);
    const float4 b = *(const float4*)(b1 + 4 * lane);
    const float4 w = *(const float4*)(W2 + 4 * lane);
    float s = fmaxf(a.x + b.x, 0.f) * w.x + fmaxf(a.y + b.y, 0.f) * w.y +
              fmaxf(a.z + b.z, 0.f) * w.z + fmaxf(a.w + b.w, 0.f) * w.w;
#pragma unroll
    for (int off = 16; off > 0; off >>= 1) s += __shfl_xor(s, off, 32);
    if (lane == 0) z[node] = s;
}

__global__ void init2(const float* __restrict__ z, const float* __restrict__ dinv,
                      float* __restrict__ a2, int n) {
    int i = blockIdx.x * blockDim.x + threadIdx.x;
    if (i < n) a2[i] = dinv[i] * dinv[i] * z[i];      // self-loop message
}

__global__ void scatter2(const int* __restrict__ edges, const float* __restrict__ dinv,
                         const float* __restrict__ z, float* __restrict__ a2, int E) {
    int e = blockIdx.x * blockDim.x + threadIdx.x;
    if (e < E) {
        int r = edges[e], c = edges[E + e];
        atomicAdd(a2 + c, dinv[r] * dinv[c] * z[r]);
    }
}

__global__ void final_sig(const float* __restrict__ a2, const float* __restrict__ b2,
                          float* __restrict__ out, int n) {
    int i = blockIdx.x * blockDim.x + threadIdx.x;
    if (i < n) {
        float t = a2[i] + b2[0];                       // relu then sigmoid:
        t = fmaxf(t, 0.f);                             // relu(agg2 + b2)
        out[i] = 1.0f / (1.0f + expf(-t));
    }
}

// ---------------------------------------------------------------------------
extern "C" void kernel_launch(void* const* d_in, const int* in_sizes, int n_in,
                              void* d_out, int out_size, void* d_ws, size_t ws_size,
                              hipStream_t stream) {
    const float* x     = (const float*)d_in[0];
    const int*   edges = (const int*)  d_in[1];
    const float* W1    = (const float*)d_in[2];
    const float* b1    = (const float*)d_in[3];
    const float* W2    = (const float*)d_in[4];
    const float* b2    = (const float*)d_in[5];
    float* out = (float*)d_out;

    const int n = in_sizes[0] / DIN;      // 100000
    const int E = in_sizes[1] / 2;        // 1600000

    // workspace layout (floats)
    float* ws   = (float*)d_ws;
    const size_t per = (size_t)n * DHID;
    float* xw   = ws;                     // n*128
    float* agg  = ws + per;               // n*128
    float* deg  = agg + per;              // n
    float* dinv = deg + n;                // n
    float* z    = dinv + n;               // n
    float* a2   = z + n;                  // n

    const int B = 256;
    fill_deg   <<<(n + B - 1) / B, B, 0, stream>>>(deg, n);
    deg_scatter<<<(E + B - 1) / B, B, 0, stream>>>(edges, deg, E);
    calc_dinv  <<<(n + B - 1) / B, B, 0, stream>>>(deg, dinv, n);

    gemm1_wmma <<<(n + 127) / 128, B, 0, stream>>>(x, W1, xw, n);
    init_agg   <<<((n * 32) + B - 1) / B, B, 0, stream>>>(xw, dinv, agg, n);
    scatter1   <<<2048, B, 0, stream>>>(edges, dinv, xw, agg, E);

    gemv2      <<<(n + 7) / 8, B, 0, stream>>>(agg, b1, W2, z, n);
    init2      <<<(n + B - 1) / B, B, 0, stream>>>(z, dinv, a2, n);
    scatter2   <<<(E + B - 1) / B, B, 0, stream>>>(edges, dinv, z, a2, E);
    final_sig  <<<(n + B - 1) / B, B, 0, stream>>>(a2, b2, out, n);
}
